// DynamicSparseRetriever_10033043603693
// MI455X (gfx1250) — matrix operand order repaired and checked by reading, not compiled, not run on hardware
//
#include <hip/hip_runtime.h>

// ---------------------------------------------------------------------------
// DynamicSparseRetriever for MI455X (gfx1250, wave32, WMMA)
//
// Roofline: context stream = 1.07 GB fp32 -> ~46us @ 23.3 TB/s is the floor.
// GEMM = 68.7 GFLOP; done as bf16 WMMA (f32 accum) so compute ~15us << memory.
// TILE_M=64 / 4 accumulators per wave keeps VGPRs ~130 -> ~8 waves/SIMD so
// LDS-fragment latency and HBM latency are hidden by occupancy.
// ---------------------------------------------------------------------------

#define BATCH 8
#define NQ    64
#define NC    32768
#define ED    1024
#define RD    128          // reduced dim == N
#define TILE_M 64          // context rows per block
#define MT     (TILE_M / 16)   // 4 M-subtiles per wave
#define CK     64          // K chunk staged in LDS
#define NCHUNK (ED / CK)   // 16
#define LDA_H  68          // padded halves per A row (conflict-free frags)
#define LDB_H  68          // padded halves per B (transposed [n][k]) row

typedef __attribute__((ext_vector_type(16))) __bf16 v16bf;
typedef __attribute__((ext_vector_type(8)))  float  v8f;

union ABFrag { unsigned int u[8]; v16bf v; };

__device__ __forceinline__ unsigned short f2bf(float x) {
  unsigned int u = __float_as_uint(x);
  unsigned int r = u + 0x7FFFu + ((u >> 16) & 1u);   // round-to-nearest-even
  return (unsigned short)(r >> 16);
}
__device__ __forceinline__ unsigned int packbf(float lo, float hi) {
  return (unsigned int)f2bf(lo) | ((unsigned int)f2bf(hi) << 16);
}
__device__ __forceinline__ unsigned int fkey(float f) {
  unsigned int u = __float_as_uint(f);
  return (u & 0x80000000u) ? ~u : (u | 0x80000000u); // orderable: bigger key = bigger float
}

// A fragment (16x32 bf16, ISA 16-bit A layout): lane row m,
// VGPR v<4: K = kt*32 + 8g + 2v + {0,1}; v>=4: K = kt*32 + 16 + 8g + 2(v-4) + {0,1}
__device__ __forceinline__ void loadAfrag(ABFrag& f, const unsigned int* LA,
                                          int m, int kt, int g) {
  const int abase = m * (LDA_H / 2) + kt * 16 + 4 * g;
  #pragma unroll
  for (int v = 0; v < 4; ++v) f.u[v] = LA[abase + v];
  #pragma unroll
  for (int v = 0; v < 4; ++v) f.u[4 + v] = LA[abase + 8 + v];
}
// B fragment (32x16 bf16): lane (g,l): col n,
// VGPR v<4: K = kt*32 + 8g + 2v + {0,1}; v>=4: K = kt*32 + 16 + 8g + 2(v-4) + {0,1}
__device__ __forceinline__ void loadBfrag(ABFrag& f, const unsigned int* LB,
                                          int n, int kt, int g) {
  const int bbase = n * (LDB_H / 2) + kt * 16 + 4 * g;
  #pragma unroll
  for (int v = 0; v < 4; ++v) f.u[v] = LB[bbase + v];
  #pragma unroll
  for (int v = 0; v < 4; ++v) f.u[4 + v] = LB[bbase + 8 + v];
}

// ---------------------------------------------------------------------------
// k_query: grid = BATCH*8 blocks. Block (b, rg) handles 8 query rows:
//   q_red rows = l2norm(q @ Wq + bq); atomically accumulates sum of normalized
//   rows into macc[b][128] and raw row sums into racc[b][1024].
// ---------------------------------------------------------------------------
__global__ __launch_bounds__(256)
void k_query(const float* __restrict__ q, const float* __restrict__ Wq,
             const float* __restrict__ bq, float* __restrict__ macc,
             float* __restrict__ racc) {
  __shared__ float lq[8 * ED];        // 32KB: 8 query rows
  __shared__ float qr[8 * RD];        // 4KB: reduced rows
  __shared__ float invn[8];
  const int tid = threadIdx.x;
  const int b  = blockIdx.x >> 3;
  const int rg = blockIdx.x & 7;
  const float* qb = q + ((size_t)b * NQ + (size_t)rg * 8) * ED;

  for (int i = tid; i < 8 * ED; i += 256) lq[i] = qb[i];
  __syncthreads();

  // raw pooled partial sums (for complexity MLP)
  for (int e = tid; e < ED; e += 256) {
    float s = 0.f;
    #pragma unroll
    for (int r = 0; r < 8; ++r) s += lq[r * ED + e];
    atomicAdd(&racc[b * ED + e], s);
  }

  // q_red = q @ Wq + bq   (8 rows x 128 cols, dots of length 1024)
  for (int i = tid; i < 8 * RD; i += 256) {
    const int r = i >> 7, c = i & 127;
    float s = bq[c];
    const float* qq = &lq[r * ED];
    for (int e = 0; e < ED; ++e) s = fmaf(qq[e], Wq[e * RD + c], s);
    qr[r * RD + c] = s;
  }
  __syncthreads();

  if (tid < 8) {
    float s = 0.f;
    for (int c = 0; c < RD; ++c) { float v = qr[tid * RD + c]; s += v * v; }
    invn[tid] = 1.f / fmaxf(sqrtf(s), 1e-12f);
  }
  __syncthreads();

  for (int i = tid; i < 8 * RD; i += 256) {
    const int r = i >> 7, c = i & 127;
    atomicAdd(&macc[b * RD + c], qr[r * RD + c] * invn[r]);
  }
}

// ---------------------------------------------------------------------------
// k_finalize: grid = BATCH blocks. q_pooled = l2norm(macc/64); complexity MLP
// -> per-batch token budget (RNE rounding matches jnp.round).
// ---------------------------------------------------------------------------
__global__ __launch_bounds__(256)
void k_finalize(const float* __restrict__ macc, const float* __restrict__ racc,
                const float* __restrict__ W1, const float* __restrict__ b1,
                const float* __restrict__ W2, const float* __restrict__ b2,
                float* __restrict__ qpool, int* __restrict__ budget) {
  __shared__ float pr[ED];
  __shared__ float red[RD];
  __shared__ float meanv[RD];
  __shared__ float hid[RD];
  const int tid = threadIdx.x;
  const int b = blockIdx.x;

  for (int e = tid; e < ED; e += 256) pr[e] = racc[b * ED + e] * (1.f / 64.f);
  if (tid < RD) {
    float m = macc[b * RD + tid] * (1.f / 64.f);
    meanv[tid] = m;
    red[tid] = m * m;
  }
  __syncthreads();
  for (int s = 64; s > 0; s >>= 1) {
    if (tid < s) red[tid] += red[tid + s];
    __syncthreads();
  }
  const float nrm = fmaxf(sqrtf(red[0]), 1e-12f);
  __syncthreads();
  if (tid < RD) qpool[b * RD + tid] = meanv[tid] / nrm;

  if (tid < RD) {   // hidden = relu(pooled_raw @ W1 + b1)
    float s = b1[tid];
    for (int e = 0; e < ED; ++e) s = fmaf(pr[e], W1[e * RD + tid], s);
    hid[tid] = fmaxf(s, 0.f);
  }
  __syncthreads();
  if (tid < RD) red[tid] = hid[tid] * W2[tid];
  __syncthreads();
  for (int s = 64; s > 0; s >>= 1) {
    if (tid < s) red[tid] += red[tid + s];
    __syncthreads();
  }
  if (tid == 0) {
    const float z  = red[0] + b2[0];
    const float cx = 1.f / (1.f + expf(-z));
    budget[b] = (int)rintf(512.f * (1.f + 0.5f * cx));
  }
}

// ---------------------------------------------------------------------------
// k_main: the WMMA kernel. grid = BATCH * (NC/TILE_M) = 4096 blocks,
// 256 threads = 8 wave32s. Wave w owns N-slice [w*16, w*16+16) across 4
// M-subtiles of 16 rows. K staged in bf16 LDS chunks of 64. A-fragment LDS
// reads are pipelined one step ahead of each v_wmma; ~130 VGPRs -> high
// occupancy hides LDS and HBM latency.
// Fused epilogue: +bc, row l2norm, dot with q_pooled -> scores.
// ---------------------------------------------------------------------------
__global__ __launch_bounds__(256)
void k_main(const float* __restrict__ ctx, const unsigned char* __restrict__ cmask,
            const float* __restrict__ Wc, const float* __restrict__ bc,
            const float* __restrict__ qpool, float* __restrict__ outScores) {
  __shared__ __align__(16) unsigned short lA[TILE_M * LDA_H];  // [m][k] bf16, 8704B
  __shared__ __align__(16) unsigned short lB[RD * LDB_H];      // [n][k] bf16, 17408B
  __shared__ float sSS[TILE_M];
  __shared__ float sDot[TILE_M];

  const int tid  = threadIdx.x;
  const int w    = tid >> 5;
  const int lane = tid & 31;
  const int g    = lane >> 4;     // half-wave group (K/M-row split per WMMA layout)
  const int l    = lane & 15;
  const int blk  = blockIdx.x;
  const int b    = blk >> 9;                   // NC/TILE_M = 512 tiles per batch
  const int row0 = (blk & 511) * TILE_M;
  const int n0   = w * 16;

  const float* A = ctx + ((size_t)b * NC + row0) * ED;
  unsigned int* LA = (unsigned int*)lA;
  unsigned int* LB = (unsigned int*)lB;

  v8f acc[MT];
  #pragma unroll
  for (int t = 0; t < MT; ++t)
    #pragma unroll
    for (int e = 0; e < 8; ++e) acc[t][e] = 0.f;

  for (int kc = 0; kc < NCHUNK; ++kc) {
    const int k0 = kc * CK;
    __syncthreads();    // previous chunk's fragment reads complete

    if (kc + 1 < NCHUNK) {  // gfx1250 global_prefetch_b8 of next A chunk
      const float* nA = A + k0 + CK;
      __builtin_prefetch(nA + (size_t)(tid >> 2) * ED + (size_t)(tid & 3) * 16, 0, 3);
    }

    // Stage A: 64 rows x 64 k, packed bf16 pairs. Coalesced float2 reads,
    // conflict-free dword LDS writes (row stride 34 dwords).
    #pragma unroll
    for (int it = 0; it < 8; ++it) {
      const int i  = tid + 256 * it;          // pair index: 64m x 32kp
      const int m  = i >> 5, kp = i & 31;
      const float2 v = *(const float2*)(A + (size_t)m * ED + k0 + 2 * kp);
      LA[m * (LDA_H / 2) + kp] = packbf(v.x, v.y);
    }
    // Stage B: Wc[k0..k0+64)[128] transposed to [n][k]. Coalesced row reads.
    #pragma unroll
    for (int it = 0; it < 16; ++it) {
      const int i  = tid + 256 * it;          // 32kp x 128n
      const int kp = i >> 7, n = i & 127;
      const float x = Wc[(size_t)(k0 + 2 * kp) * RD + n];
      const float y = Wc[(size_t)(k0 + 2 * kp + 1) * RD + n];
      LB[n * (LDB_H / 2) + kp] = packbf(x, y);
    }
    __syncthreads();

    // Software-pipelined compute: both B fragments and the next A fragment
    // are resident/in-flight before each v_wmma issues.
    ABFrag bfr[CK / 32];
    loadBfrag(bfr[0], LB, n0 + l, 0, g);
    ABFrag cur, nxt;
    loadAfrag(cur, LA, l, 0, g);              // t=0 of kt=0
    loadBfrag(bfr[1], LB, n0 + l, 1, g);

    #pragma unroll
    for (int kt = 0; kt < CK / 32; ++kt) {
      #pragma unroll
      for (int t = 0; t < MT; ++t) {
        if (t < MT - 1)            loadAfrag(nxt, LA, (t + 1) * 16 + l, kt, g);
        else if (kt + 1 < CK / 32) loadAfrag(nxt, LA, l, kt + 1, g);
        acc[t] = __builtin_amdgcn_wmma_f32_16x16x32_bf16(
            false, cur.v, false, bfr[kt].v, (short)0, acc[t], false, false);
        cur = nxt;
      }
    }
  }

  // Epilogue: c_red = acc + bc; per-row sum(x^2) and sum(x*q_pooled)
  __syncthreads();
  if (tid < TILE_M) sSS[tid] = 0.f;
  else if (tid < 2 * TILE_M) sDot[tid - TILE_M] = 0.f;
  __syncthreads();

  const float qpv = qpool[b * RD + n0 + l];
  const float bcv = bc[n0 + l];
  #pragma unroll
  for (int t = 0; t < MT; ++t) {
    #pragma unroll
    for (int r = 0; r < 8; ++r) {
      // C/D layout: VGPR r -> row = t*16 + r + 8g, col = n0 + l
      float val = acc[t][r] + bcv;
      float a2 = val * val;
      float dd = val * qpv;
      #pragma unroll
      for (int m = 1; m < 16; m <<= 1) {   // reduce over 16 cols (half-wave)
        a2 += __shfl_xor(a2, m, 16);
        dd += __shfl_xor(dd, m, 16);
      }
      if (l == 0) {
        const int row = t * 16 + r + 8 * g;
        atomicAdd(&sSS[row], a2);
        atomicAdd(&sDot[row], dd);
      }
    }
  }
  __syncthreads();

  if (tid < TILE_M) {
    const float nrm = sqrtf(sSS[tid]);
    float sc = sDot[tid] / fmaxf(nrm, 1e-12f);
    const size_t gi = (size_t)b * NC + row0 + tid;
    if (!cmask[gi]) sc = -__builtin_inff();
    outScores[gi] = sc;
  }
}

// ---------------------------------------------------------------------------
// k_select: grid = BATCH blocks. Exact 4-pass radix select of the k-th largest
// key, then index-ordered tie-break -> bit-exact match of
// argsort(argsort(-scores)) < budget (stable descending sort).
// ---------------------------------------------------------------------------
__global__ __launch_bounds__(256)
void k_select(const float* __restrict__ scores, const unsigned char* __restrict__ cmask,
              const int* __restrict__ budget, float* __restrict__ outMask) {
  __shared__ unsigned int hist[256];
  __shared__ int sh_msum;
  __shared__ unsigned int sh_dig;
  __shared__ int sh_rem;
  __shared__ unsigned int wcnt[8];
  __shared__ unsigned int woff[8];
  __shared__ unsigned int runEq;

  const int tid = threadIdx.x;
  const int b = blockIdx.x;
  const float* sc = scores + (size_t)b * NC;
  const unsigned char* mk = cmask + (size_t)b * NC;

  // budget clamp by valid-token count
  int cnt = 0;
  for (int i = tid; i < NC; i += 256) cnt += mk[i] ? 1 : 0;
  if (tid == 0) sh_msum = 0;
  __syncthreads();
  atomicAdd(&sh_msum, cnt);
  __syncthreads();
  int k = budget[b];
  if (k > sh_msum) k = sh_msum;
  if (k > NC) k = NC;

  // radix select (MSB -> LSB) on orderable keys
  unsigned int cur = 0, curMask = 0;
  int rem = k;
  for (int p = 3; p >= 0; --p) {
    hist[tid] = 0;   // 256 threads, 256 bins
    __syncthreads();
    for (int i = tid; i < NC; i += 256) {
      const unsigned int key = fkey(sc[i]);
      if ((key & curMask) == cur)
        atomicAdd(&hist[(key >> (8 * p)) & 255u], 1u);
    }
    __syncthreads();
    if (tid == 0) {
      int r = rem;
      unsigned int d = 0;
      for (int bin = 255; bin >= 0; --bin) {
        const int c2 = (int)hist[bin];
        if (r <= c2) { d = (unsigned int)bin; break; }
        r -= c2;
      }
      sh_dig = d;
      sh_rem = r;
    }
    __syncthreads();
    cur     |= sh_dig << (8 * p);
    curMask |= 255u   << (8 * p);
    rem = sh_rem;
    __syncthreads();
  }
  const unsigned int T = cur;
  const int Rm = rem;   // number of T-ties to take (first Rm by index)

  if (tid == 0) runEq = 0;
  __syncthreads();
  const int lane = tid & 31, w = tid >> 5;
  for (int base = 0; base < NC; base += 256) {
    const int i = base + tid;
    const unsigned int key = fkey(sc[i]);
    const bool gt = key > T;
    const bool eq = (key == T);
    const unsigned int bal = (unsigned int)__ballot((int)eq);   // wave32
    const unsigned int pre = __popc(bal & ((1u << lane) - 1u));
    if (lane == 0) wcnt[w] = __popc(bal);
    __syncthreads();
    if (tid == 0) {
      unsigned int o = runEq;
      #pragma unroll
      for (int ww = 0; ww < 8; ++ww) { woff[ww] = o; o += wcnt[ww]; }
      runEq = o;
    }
    __syncthreads();
    const unsigned int tieIdx = woff[w] + pre;
    const bool sel = gt || (eq && (int)tieIdx < Rm);
    outMask[(size_t)b * NC + i] = sel ? 1.f : 0.f;
    __syncthreads();
  }
}

// ---------------------------------------------------------------------------
extern "C" void kernel_launch(void* const* d_in, const int* in_sizes, int n_in,
                              void* d_out, int out_size, void* d_ws, size_t ws_size,
                              hipStream_t stream) {
  (void)in_sizes; (void)n_in; (void)out_size; (void)ws_size;
  const float* q    = (const float*)d_in[0];
  const float* ctx  = (const float*)d_in[1];
  const unsigned char* cm = (const unsigned char*)d_in[2];  // jnp bool = 1 byte
  const float* Wq = (const float*)d_in[3];
  const float* bq = (const float*)d_in[4];
  const float* Wc = (const float*)d_in[5];
  const float* bc = (const float*)d_in[6];
  const float* W1 = (const float*)d_in[7];
  const float* b1 = (const float*)d_in[8];
  const float* W2 = (const float*)d_in[9];
  const float* b2 = (const float*)d_in[10];

  float* ws    = (float*)d_ws;
  float* macc  = ws;                         // 8*128
  float* racc  = ws + 1024;                  // 8*1024
  float* qpool = ws + 1024 + 8192;           // 8*128
  int*   bud   = (int*)(ws + 1024 + 8192 + 1024);  // 8

  float* outMask   = (float*)d_out;                        // [8,32768]
  float* outScores = (float*)d_out + (size_t)BATCH * NC;   // [8,32768]

  hipMemsetAsync(d_ws, 0, (1024 + 8192 + 1024 + 8) * sizeof(float), stream);
  k_query   <<<BATCH * 8, 256, 0, stream>>>(q, Wq, bq, macc, racc);
  k_finalize<<<BATCH,     256, 0, stream>>>(macc, racc, W1, b1, W2, b2, qpool, bud);
  k_main    <<<BATCH * (NC / TILE_M), 256, 0, stream>>>(ctx, cm, Wc, bc, qpool, outScores);
  k_select  <<<BATCH,     256, 0, stream>>>(outScores, cm, bud, outMask);
}
